// OcclusionRecoveryNetwork_28020366639427
// MI455X (gfx1250) — matrix operand order repaired and checked by reading, not compiled, not run on hardware
//
#include <hip/hip_runtime.h>

// ---------------------------------------------------------------------------
// OcclusionRecoveryNetwork for MI455X (gfx1250, wave32, WMMA 16x16x32 f16)
// Pipeline:
//   0) f32->f16 weight conversion, patch gather (im2col for 8x8 stride-8 conv)
//   1) embed GEMM   : Xp(25088x192) @ conv_w^T -> emb f16 (both frames)
//   2) q GEMM       : cur_emb @ q_w^T, *SCALE, store (b,h,n,hd) f16
//   3) kv GEMM      : prev_emb @ kv_w^T, K -> (b,h,n,hd), V -> (b,h,hd,n) f16
//   4) flash attention: per-wave 16-query tile, 25 key-blocks of 32,
//      online softmax w/ register-resident row stats, LDS bias-table gather
//   5) proj GEMM    : attn_out @ proj_w^T + b -> f32 output
// GEMM cores are software-pipelined (register double buffering) and the
// load->WMMA phase order is pinned with __builtin_amdgcn_sched_barrier so the
// pressure-aware scheduler cannot sink next-step loads below the current WMMAs.
// ---------------------------------------------------------------------------

typedef __attribute__((ext_vector_type(16))) _Float16 v16h;
typedef __attribute__((ext_vector_type(8)))  _Float16 v8h;
typedef __attribute__((ext_vector_type(8)))  float    v8f;

#define B_     16
#define CIN    3
#define IMG    224
#define P_     8
#define D_     256
#define NHD    8
#define HD_    32
#define GRID_  28
#define NSEQ   784                 // 28*28 patches
#define KPATCH 192                 // 3*8*8
#define MROWS  (B_ * NSEQ)         // 12544
#define M2     (2 * MROWS)         // 25088 (both frames)
#define NPAD   800                 // padded key stride for V^T (25*32)
#define TT     61
#define TBL    (TT * TT)           // 3721
#define SCALE_ 0.17677669529663687f  // 32^-0.5

#if defined(__has_builtin)
#if __has_builtin(__builtin_amdgcn_sched_barrier)
#define SCHED_FENCE() __builtin_amdgcn_sched_barrier(0)
#endif
#endif
#ifndef SCHED_FENCE
#define SCHED_FENCE()
#endif

// ---------------------------------------------------------------------------
__device__ __forceinline__ v8f wmma_f16(v16h a, v16h b, v8f c) {
  // (neg_a, A, neg_b, B, c_mod, C, reuse_a, reuse_b)
  return __builtin_amdgcn_wmma_f32_16x16x32_f16(false, a, false, b,
                                                (short)0, c, false, false);
}

// C(16x64) += A(16xK) * W^T tile, K a compile-time constant, fully unrolled
// and double-buffered: step kk+32 loads are pinned (sched_barrier) above step
// kk's WMMAs so they stay in flight across the matrix math.
// A-layout (16-bit 16x32): lane row = lane&15; lane<16 holds K {kk+0..15},
// lane>=16 holds K {kk+16..31}.
// B-layout (32x16): lane<16 = column n, rows kk..kk+15; lane>=16 rows kk+16..31.
template <int K>
__device__ __forceinline__ void gemm_core_16x64(
    const _Float16* __restrict__ A, int lda,
    const _Float16* __restrict__ Wt, int ldw,
    v8f acc[4]) {
  const int lane = threadIdx.x & 31;
  const int l16  = lane & 15;
  const int half = lane >> 4;
  const _Float16* arow = A + (size_t)l16 * lda + half * 16;
  const _Float16* wrow = Wt + (size_t)l16 * ldw + half * 16;

  v16h a_cur, b_cur[4];
  *(v8h*)&a_cur       = *(const v8h*)(arow);
  *((v8h*)&a_cur + 1) = *(const v8h*)(arow + 8);
#pragma unroll
  for (int c = 0; c < 4; ++c)
    b_cur[c] = *(const v16h*)(wrow + (size_t)c * 16 * ldw);

#pragma unroll
  for (int kk = 0; kk < K; kk += 32) {
    v16h a_nxt, b_nxt[4];
    if (kk + 32 < K) {  // issue next-step loads before current WMMAs
      *(v8h*)&a_nxt       = *(const v8h*)(arow + kk + 32);
      *((v8h*)&a_nxt + 1) = *(const v8h*)(arow + kk + 40);
#pragma unroll
      for (int c = 0; c < 4; ++c)
        b_nxt[c] = *(const v16h*)(wrow + (size_t)c * 16 * ldw + kk + 32);
    }
    SCHED_FENCE();  // loads above may not sink below; WMMAs may not hoist above
#pragma unroll
    for (int c = 0; c < 4; ++c)
      acc[c] = wmma_f16(a_cur, b_cur[c], acc[c]);
    SCHED_FENCE();
    if (kk + 32 < K) {
      a_cur = a_nxt;
#pragma unroll
      for (int c = 0; c < 4; ++c) b_cur[c] = b_nxt[c];
    }
  }
}

// ---------------------------------------------------------------------------
__global__ void f32_to_f16_kernel(const float* __restrict__ src,
                                  _Float16* __restrict__ dst, int n) {
  int i = blockIdx.x * blockDim.x + threadIdx.x;
  if (i < n) dst[i] = (_Float16)src[i];
}

// im2col: one block per patch-row (blockDim.x == 192 == KPATCH)
__global__ void patch_gather_kernel(const float* __restrict__ cur,
                                    const float* __restrict__ prev,
                                    _Float16* __restrict__ Xp) {
  int k = threadIdx.x;       // 0..191
  int r = blockIdx.x;        // 0..25087
  const float* src = (r < MROWS) ? cur : prev;
  int rr = (r < MROWS) ? r : r - MROWS;
  int b = rr / NSEQ, n = rr % NSEQ;
  int py = n / GRID_, px = n % GRID_;
  int c  = k >> 6, k2 = k & 63, ky = k2 >> 3, kx = k2 & 7;
  float v = src[((size_t)(b * CIN + c) * IMG + py * P_ + ky) * IMG + px * P_ + kx];
  Xp[(size_t)r * KPATCH + k] = (_Float16)v;
}

// ---------------------------------------------------------------------------
__global__ __launch_bounds__(256, 2) void embed_gemm_kernel(
    const _Float16* __restrict__ Xp, const _Float16* __restrict__ Wc,
    const float* __restrict__ bias, _Float16* __restrict__ emb) {
  int wave = (blockIdx.x * blockDim.x + threadIdx.x) >> 5;
  int tn = wave & 3, tm = wave >> 2;
  if (tm >= M2 / 16) return;
  v8f acc[4] = {};
  gemm_core_16x64<KPATCH>(Xp + (size_t)tm * 16 * KPATCH, KPATCH,
                          Wc + (size_t)tn * 64 * KPATCH, KPATCH, acc);
  int lane = threadIdx.x & 31, l16 = lane & 15, half = lane >> 4;
#pragma unroll
  for (int c = 0; c < 4; ++c) {
    int n = tn * 64 + c * 16 + l16;
    float bv = bias[n];
#pragma unroll
    for (int r = 0; r < 8; ++r) {
      int m = tm * 16 + 8 * half + r;
      emb[(size_t)m * D_ + n] = (_Float16)(acc[c][r] + bv);
    }
  }
}

__global__ __launch_bounds__(256, 2) void q_gemm_kernel(
    const _Float16* __restrict__ emb, const _Float16* __restrict__ Wq,
    const float* __restrict__ qb, _Float16* __restrict__ q16) {
  int wave = (blockIdx.x * blockDim.x + threadIdx.x) >> 5;
  int tn = wave & 3, tm = wave >> 2;
  if (tm >= MROWS / 16) return;
  v8f acc[4] = {};
  gemm_core_16x64<D_>(emb + (size_t)tm * 16 * D_, D_,
                      Wq + (size_t)tn * 64 * D_, D_, acc);
  int lane = threadIdx.x & 31, l16 = lane & 15, half = lane >> 4;
#pragma unroll
  for (int c = 0; c < 4; ++c) {
    int n = tn * 64 + c * 16 + l16;
    int h = n >> 5, hd = n & 31;
    float bv = qb[n];
#pragma unroll
    for (int r = 0; r < 8; ++r) {
      int m = tm * 16 + 8 * half + r;
      int b = m / NSEQ, ns = m % NSEQ;
      q16[((size_t)(b * NHD + h) * NSEQ + ns) * HD_ + hd] =
          (_Float16)((acc[c][r] + bv) * SCALE_);
    }
  }
}

__global__ __launch_bounds__(256, 2) void kv_gemm_kernel(
    const _Float16* __restrict__ emb, const _Float16* __restrict__ Wkv,
    const float* __restrict__ kvb, _Float16* __restrict__ k16,
    _Float16* __restrict__ vT16) {
  int wave = (blockIdx.x * blockDim.x + threadIdx.x) >> 5;
  int tn = wave & 7, tm = wave >> 3;
  if (tm >= MROWS / 16) return;
  v8f acc[4] = {};
  gemm_core_16x64<D_>(emb + (size_t)(MROWS + tm * 16) * D_, D_,   // prev frame
                      Wkv + (size_t)tn * 64 * D_, D_, acc);
  int lane = threadIdx.x & 31, l16 = lane & 15, half = lane >> 4;
#pragma unroll
  for (int c = 0; c < 4; ++c) {
    int n = tn * 64 + c * 16 + l16;
    float bv = kvb[n];
#pragma unroll
    for (int r = 0; r < 8; ++r) {
      int m = tm * 16 + 8 * half + r;
      int b = m / NSEQ, ns = m % NSEQ;
      float val = acc[c][r] + bv;
      if (n < D_) {                         // K: (b,h,n,hd)
        int h = n >> 5, hd = n & 31;
        k16[((size_t)(b * NHD + h) * NSEQ + ns) * HD_ + hd] = (_Float16)val;
      } else {                              // V transposed: (b,h,hd,n) stride NPAD
        int d = n - D_, h = d >> 5, hd = d & 31;
        vT16[((size_t)(b * NHD + h) * HD_ + hd) * NPAD + ns] = (_Float16)val;
      }
    }
  }
}

// ---------------------------------------------------------------------------
// Flash attention. Block = 8 waves; wave = one 16-query tile of one (b,h).
// 25 key-blocks of 32 (last half-masked; padded V stride makes tail loads safe,
// multiplied by p==0). Row stats replicated per 16-lane half via shfl_xor(,16).
// K and V tiles for a block are all issued up front so they overlap the
// softmax VALU work; P round-trips through LDS to switch C-layout -> A-layout.
__global__ __launch_bounds__(256, 2) void flash_attn_kernel(
    const _Float16* __restrict__ q16, const _Float16* __restrict__ k16,
    const _Float16* __restrict__ vT16, const float* __restrict__ bias_table,
    _Float16* __restrict__ attn_out) {
  __shared__ float    s_bias[TBL];
  __shared__ _Float16 s_p[8][16 * 32];

  const int h = blockIdx.y, b = blockIdx.z;
  for (int i = threadIdx.x; i < TBL; i += blockDim.x)
    s_bias[i] = bias_table[(size_t)i * NHD + h];
  __syncthreads();

  const int wave = threadIdx.x >> 5;
  const int qt = blockIdx.x * 8 + wave;
  if (qt >= 49) return;
  const int lane = threadIdx.x & 31, l16 = lane & 15, half = lane >> 4;

  const _Float16* qbase = q16  + (size_t)(b * NHD + h) * NSEQ * HD_;
  const _Float16* kbase = k16  + (size_t)(b * NHD + h) * NSEQ * HD_;
  const _Float16* vbase = vT16 + (size_t)(b * NHD + h) * HD_ * NPAD;
  _Float16* sp = s_p[wave];

  // Q tile in A-layout (row = l16, K-halves by lane half)
  v16h aq;
  {
    const _Float16* qr = qbase + (size_t)(qt * 16 + l16) * HD_ + half * 16;
    *(v8h*)&aq       = *(const v8h*)qr;
    *((v8h*)&aq + 1) = *(const v8h*)(qr + 8);
  }

  // per-row query coords for the 8 rows this lane's accumulators cover
  int qy[8], qx[8];
#pragma unroll
  for (int r = 0; r < 8; ++r) {
    int qq = qt * 16 + 8 * half + r;
    qy[r] = qq / GRID_; qx[r] = qq % GRID_;
  }

  v8f o0 = {}, o1 = {}, m_run, l_run;
#pragma unroll
  for (int r = 0; r < 8; ++r) { m_run[r] = -1e30f; l_run[r] = 0.f; }

  const v8f zero = {};
  for (int kb = 0; kb < 25; ++kb) {
    const bool fullblk = (kb < 24);
    const int key0 = kb * 32;
    if (kb + 1 < 25) {  // CDNA5 global_prefetch_b8 for next K/V tiles
      __builtin_prefetch(kbase + (size_t)(key0 + 32) * HD_, 0, 1);
      __builtin_prefetch(vbase + key0 + 32, 0, 1);
    }

    // issue ALL global loads for this block first (K chunks + V chunks) so
    // they overlap the softmax VALU work below
    const int keyA = key0 + l16;
    v16h bk0 = *(const v16h*)(kbase + (size_t)keyA * HD_ + half * 16);
    v16h bk1;
    if (fullblk)
      bk1 = *(const v16h*)(kbase + (size_t)(keyA + 16) * HD_ + half * 16);
    v16h bv0 = *(const v16h*)(vbase + (size_t)(l16)      * NPAD + key0 + half * 16);
    v16h bv1 = *(const v16h*)(vbase + (size_t)(16 + l16) * NPAD + key0 + half * 16);

    // S = Q * K^T  (two 16-key chunks)
    v8f s0 = wmma_f16(aq, bk0, zero);
    v8f s1 = zero;
    if (fullblk) s1 = wmma_f16(aq, bk1, zero);

    // + relative position bias (gathered from LDS table slice)
    {
      int ky = keyA / GRID_, kx = keyA % GRID_;
#pragma unroll
      for (int r = 0; r < 8; ++r)
        s0[r] += s_bias[(qy[r] - ky + 30) * TT + (qx[r] - kx + 30)];
      if (fullblk) {
        int kB = keyA + 16, ky1 = kB / GRID_, kx1 = kB % GRID_;
#pragma unroll
        for (int r = 0; r < 8; ++r)
          s1[r] += s_bias[(qy[r] - ky1 + 30) * TT + (qx[r] - kx1 + 30)];
      }
    }

    // online softmax update
#pragma unroll
    for (int r = 0; r < 8; ++r) {
      float v = fullblk ? fmaxf(s0[r], s1[r]) : s0[r];
#pragma unroll
      for (int o = 8; o >= 1; o >>= 1) v = fmaxf(v, __shfl_xor(v, o, 16));
      float nm = fmaxf(m_run[r], v);
      float sc = __expf(m_run[r] - nm);
      m_run[r] = nm;
      o0[r] *= sc; o1[r] *= sc;
      float p0 = __expf(s0[r] - nm);
      float p1 = fullblk ? __expf(s1[r] - nm) : 0.f;
      float rs = p0 + p1;
#pragma unroll
      for (int o = 8; o >= 1; o >>= 1) rs += __shfl_xor(rs, o, 16);
      l_run[r] = l_run[r] * sc + rs;
      // stage P (C-layout -> row-major f16 in LDS) for the PV A-operand
      int m = 8 * half + r;
      sp[m * 32 + l16]      = (_Float16)p0;
      sp[m * 32 + 16 + l16] = (_Float16)p1;
    }
    asm volatile("s_wait_dscnt 0x0" ::: "memory");  // cross-lane LDS visibility

    // reload P in A-layout
    v16h pa;
    {
      const _Float16* pr = sp + l16 * 32 + half * 16;
      *(v8h*)&pa       = *(const v8h*)pr;
      *((v8h*)&pa + 1) = *(const v8h*)(pr + 8);
    }

    // O += P * V   (V tiles already in registers)
    o0 = wmma_f16(pa, bv0, o0);
    o1 = wmma_f16(pa, bv1, o1);
  }

  // normalize and store attn_out as (b, q, h*HD+hd) f16
  _Float16* obase = attn_out + ((size_t)b * NSEQ + qt * 16) * D_ + h * HD_;
#pragma unroll
  for (int r = 0; r < 8; ++r) {
    int m = 8 * half + r;
    float inv = 1.0f / l_run[r];
    obase[(size_t)m * D_ + l16]      = (_Float16)(o0[r] * inv);
    obase[(size_t)m * D_ + 16 + l16] = (_Float16)(o1[r] * inv);
  }
}

// ---------------------------------------------------------------------------
__global__ __launch_bounds__(256, 2) void proj_gemm_kernel(
    const _Float16* __restrict__ X, const _Float16* __restrict__ Wp,
    const float* __restrict__ pb, float* __restrict__ out) {
  int wave = (blockIdx.x * blockDim.x + threadIdx.x) >> 5;
  int tn = wave & 3, tm = wave >> 2;
  if (tm >= MROWS / 16) return;
  v8f acc[4] = {};
  gemm_core_16x64<D_>(X + (size_t)tm * 16 * D_, D_,
                      Wp + (size_t)tn * 64 * D_, D_, acc);
  int lane = threadIdx.x & 31, l16 = lane & 15, half = lane >> 4;
#pragma unroll
  for (int c = 0; c < 4; ++c) {
    int n = tn * 64 + c * 16 + l16;
    float bv = pb[n];
#pragma unroll
    for (int r = 0; r < 8; ++r) {
      int m = tm * 16 + 8 * half + r;
      out[(size_t)m * D_ + n] = acc[c][r] + bv;
    }
  }
}

// ---------------------------------------------------------------------------
extern "C" void kernel_launch(void* const* d_in, const int* in_sizes, int n_in,
                              void* d_out, int out_size, void* d_ws, size_t ws_size,
                              hipStream_t stream) {
  const float* cur_frame  = (const float*)d_in[0];
  const float* prev_frame = (const float*)d_in[1];
  const float* conv_w     = (const float*)d_in[2];
  const float* conv_b     = (const float*)d_in[3];
  const float* q_w        = (const float*)d_in[4];
  const float* q_b        = (const float*)d_in[5];
  const float* kv_w       = (const float*)d_in[6];
  const float* kv_b       = (const float*)d_in[7];
  const float* proj_w     = (const float*)d_in[8];
  const float* proj_b     = (const float*)d_in[9];
  const float* bias_table = (const float*)d_in[10];
  float* out = (float*)d_out;

  char* ws = (char*)d_ws;
  size_t off = 0;
  auto alloc = [&](size_t bytes) -> void* {
    off = (off + 255) & ~(size_t)255;
    void* p = ws + off;
    off += bytes;
    return p;
  };
  _Float16* wconv16 = (_Float16*)alloc((size_t)D_ * KPATCH * 2);
  _Float16* wq16    = (_Float16*)alloc((size_t)D_ * D_ * 2);
  _Float16* wkv16   = (_Float16*)alloc((size_t)2 * D_ * D_ * 2);
  _Float16* wproj16 = (_Float16*)alloc((size_t)D_ * D_ * 2);
  _Float16* Xp      = (_Float16*)alloc((size_t)M2 * KPATCH * 2);
  _Float16* emb     = (_Float16*)alloc((size_t)M2 * D_ * 2);
  _Float16* q16     = (_Float16*)alloc((size_t)B_ * NHD * NSEQ * HD_ * 2);
  _Float16* k16     = (_Float16*)alloc((size_t)B_ * NHD * NSEQ * HD_ * 2);
  _Float16* vT16    = (_Float16*)alloc((size_t)B_ * NHD * HD_ * NPAD * 2);
  _Float16* att16   = (_Float16*)alloc((size_t)MROWS * D_ * 2);
  (void)ws_size; (void)in_sizes; (void)n_in; (void)out_size;

  f32_to_f16_kernel<<<(D_ * KPATCH + 255) / 256, 256, 0, stream>>>(conv_w, wconv16, D_ * KPATCH);
  f32_to_f16_kernel<<<(D_ * D_ + 255) / 256, 256, 0, stream>>>(q_w, wq16, D_ * D_);
  f32_to_f16_kernel<<<(2 * D_ * D_ + 255) / 256, 256, 0, stream>>>(kv_w, wkv16, 2 * D_ * D_);
  f32_to_f16_kernel<<<(D_ * D_ + 255) / 256, 256, 0, stream>>>(proj_w, wproj16, D_ * D_);

  patch_gather_kernel<<<M2, KPATCH, 0, stream>>>(cur_frame, prev_frame, Xp);

  embed_gemm_kernel<<<(M2 / 16) * 4 / 8, 256, 0, stream>>>(Xp, wconv16, conv_b, emb);
  q_gemm_kernel<<<(MROWS / 16) * 4 / 8, 256, 0, stream>>>(emb, wq16, q_b, q16);
  kv_gemm_kernel<<<(MROWS / 16) * 8 / 8, 256, 0, stream>>>(emb, wkv16, kv_b, k16, vT16);

  dim3 fgrid(7, NHD, B_);  // ceil(49/8) qtile-groups, heads, batch
  flash_attn_kernel<<<fgrid, 256, 0, stream>>>(q16, k16, vT16, bias_table, att16);

  proj_gemm_kernel<<<(MROWS / 16) * 4 / 8, 256, 0, stream>>>(att16, wproj16, proj_b, out);
}
